// nTSNTM_60936995995819
// MI455X (gfx1250) — matrix operand order repaired and checked by reading, not compiled, not run on hardware
//
#include <hip/hip_runtime.h>
#include <math.h>

// ---------------- problem dims ----------------
constexpr int Vd  = 50000;
constexpr int Hd  = 512;
constexpr int Td  = 64;
constexpr int T2d = 16;
constexpr int Bd  = 512;
constexpr int KCAT = 96;              // 64 + 16 + 1 padded to 3*32
constexpr int SPLIT = 8;              // split-K slices for the big GEMM
constexpr int RECON_GY = 125;         // v-tile groups in fused_recon (125*25*16 = 50000)
constexpr float PRIOR_A = 1.0f;
constexpr float PRIOR_B = 5.0f;
constexpr float LOG_BETA_PRIOR = -1.6094379124f; // lgamma(1)+lgamma(5)-lgamma(6)

typedef __attribute__((ext_vector_type(16))) __bf16 v16bf;
typedef __attribute__((ext_vector_type(8)))  float  v8f;

// ---------------- WMMA fragment loaders ----------------
// A (16x32 bf16): lane L holds row (L&15); halves 0..7 <- K base..base+7,
// halves 8..15 <- K base+16..base+23, base = k0 + (L<16?0:8).
__device__ __forceinline__ v16bf afrag_f32(const float* __restrict__ row,
                                           int base, int K, bool rowOk, bool full) {
  v16bf f;
  if (full && rowOk) {
#pragma unroll
    for (int e = 0; e < 8; e++) f[e]     = (__bf16)row[base + e];
#pragma unroll
    for (int e = 0; e < 8; e++) f[8 + e] = (__bf16)row[base + 16 + e];
  } else {
#pragma unroll
    for (int e = 0; e < 8; e++)  { int k = base + e;      f[e]     = (__bf16)((rowOk && k < K) ? row[k] : 0.f); }
#pragma unroll
    for (int e = 0; e < 8; e++)  { int k = base + 16 + e; f[8 + e] = (__bf16)((rowOk && k < K) ? row[k] : 0.f); }
  }
  return f;
}

// B (32x16): lane L holds column n0+(L&15); halves 0..7 <- K kb..kb+7,
// halves 8..15 <- K kb+16..kb+23, kb = k0 + (L<16?0:8) (row-major B, ld = N).
__device__ __forceinline__ v16bf bfrag_f32(const float* __restrict__ B, int ldb,
                                           int k, int n0, int K, bool full) {
  v16bf f;
  const int l = (int)threadIdx.x & 31;
  const int kb = k + ((l < 16) ? 0 : 8);
  const float* p = B + n0 + (l & 15);
#pragma unroll
  for (int e = 0; e < 16; e++) {
    int kk = kb + (e < 8 ? e : e + 8);
    f[e] = (__bf16)((full || kk < K) ? p[(size_t)kk * ldb] : 0.f);
  }
  return f;
}

__device__ __forceinline__ v16bf bfrag_bf16(const __bf16* __restrict__ B, int ldb,
                                            int k, int n0, int K, bool full) {
  v16bf f;
  const int l = (int)threadIdx.x & 31;
  const int kb = k + ((l < 16) ? 0 : 8);
  const __bf16* p = B + n0 + (l & 15);
#pragma unroll
  for (int e = 0; e < 16; e++) {
    int kk = kb + (e < 8 ? e : e + 8);
    f[e] = (full || kk < K) ? p[(size_t)kk * ldb] : (__bf16)0.f;
  }
  return f;
}

// ---------------- generic WMMA GEMM ----------------
// One wave computes a 16(M) x 64(N) strip: 4 accumulators reuse one A fragment,
// all 5 fragment loads issued before the 4 back-to-back WMMAs; prefetch ahead on K.
// A: f32 row-major [M,K]; B: row-major [K,N] (f32 or bf16).
// SPLITK: blockIdx.z selects a K slice; raw partial written to C + z*M*N (no bias/act).
// ACT (non-split only): 0 = none, 1 = tanh, 2 = softplus
template <int ACT, bool BBF16, bool SPLITK>
__global__ void wmma_gemm(const float* __restrict__ A, const void* __restrict__ Bv,
                          const float* __restrict__ bias, float* __restrict__ C,
                          int M, int N, int K, int kIterPerSlice) {
  const int lane = (int)threadIdx.x;          // 0..31 (wave32)
  const int n0 = blockIdx.x * 64;
  const int m0 = blockIdx.y * 16;
  const int rc = lane & 15;
  const int halfOff = (lane < 16) ? 0 : 8;
  const int mA = m0 + rc;
  const bool aOk = (mA < M);
  const float* arow = A + (size_t)mA * K;

  bool nOk[4];
#pragma unroll
  for (int j = 0; j < 4; j++) nOk[j] = (n0 + j * 16) < N;   // N % 16 == 0 -> all-or-nothing

  const int kIterTotal = (K + 31) / 32;
  int it0 = 0, itEnd = kIterTotal;
  if (SPLITK) {
    it0 = blockIdx.z * kIterPerSlice;
    itEnd = it0 + kIterPerSlice;
    if (itEnd > kIterTotal) itEnd = kIterTotal;
  }

  v8f acc[4] = {v8f{}, v8f{}, v8f{}, v8f{}};

  for (int it = it0; it < itEnd; it++) {
    const int k0 = it * 32;
    const bool full = (k0 + 32 <= K);
    // ---- issue all fragment loads for this k-step ----
    v16bf af = afrag_f32(arow, k0 + halfOff, K, aOk, full);
    v16bf bf[4];
#pragma unroll
    for (int j = 0; j < 4; j++) {
      if (nOk[j]) {
        if (BBF16) bf[j] = bfrag_bf16((const __bf16*)Bv, N, k0, n0 + j * 16, K, full);
        else       bf[j] = bfrag_f32 ((const float *)Bv, N, k0, n0 + j * 16, K, full);
      }
    }
    // ---- prefetch a future K chunk while the WMMAs run ----
    {
      const int kpf = k0 + 128;
      if (kpf + lane < K) {
        if (BBF16) __builtin_prefetch((const __bf16*)Bv + (size_t)(kpf + lane) * N + n0, 0, 1);
        else       __builtin_prefetch((const float *)Bv + (size_t)(kpf + lane) * N + n0, 0, 1);
      }
      if (aOk && (kpf + halfOff) < K) __builtin_prefetch(arow + kpf + halfOff, 0, 1);
    }
    // ---- 4 back-to-back WMMAs sharing one A fragment ----
#pragma unroll
    for (int j = 0; j < 4; j++)
      if (nOk[j])
        acc[j] = __builtin_amdgcn_wmma_f32_16x16x32_bf16(false, af, false, bf[j],
                                                         (short)0, acc[j], false, false);
  }

  float* Cb = SPLITK ? (C + (size_t)blockIdx.z * M * N) : C;
#pragma unroll
  for (int j = 0; j < 4; j++) {
    if (!nOk[j]) continue;
    const int nOut = n0 + j * 16 + rc;
    const float bv = (!SPLITK && bias) ? bias[nOut] : 0.f;
#pragma unroll
    for (int r = 0; r < 8; r++) {
      const int m = m0 + r + ((lane < 16) ? 0 : 8);
      if (m < M) {
        float v = acc[j][r] + bv;
        if (!SPLITK) {
          if (ACT == 1) v = tanhf(v);
          if (ACT == 2) v = (v > 20.f) ? v : log1pf(expf(v));
        }
        Cb[(size_t)m * N + nOut] = v;
      }
    }
  }
}

// deterministic split-K reduce + bias + activation
template <int ACT>
__global__ void splitk_reduce(const float* __restrict__ part, const float* __restrict__ bias,
                              float* __restrict__ C, int MN, int N, int S) {
  int i = blockIdx.x * blockDim.x + threadIdx.x;
  if (i >= MN) return;
  float s = 0.f;
  for (int sl = 0; sl < S; sl++) s += part[(size_t)sl * MN + i];
  if (bias) s += bias[i % N];
  if (ACT == 1) s = tanhf(s);
  if (ACT == 2) s = (s > 20.f) ? s : log1pf(expf(s));
  C[i] = s;
}

// ---------------- small helper kernels ----------------
// word_vec [V,H] -> word_vecT bf16 [H,V] (coalesced writes)
__global__ void wv_transpose(const float* __restrict__ wv, __bf16* __restrict__ wvT) {
  size_t idx = (size_t)blockIdx.x * blockDim.x + threadIdx.x;
  if (idx >= (size_t)Hd * Vd) return;
  size_t h = idx / Vd, v = idx % Vd;
  wvT[idx] = (__bf16)wv[v * Hd + h];
}

// per-b Gaussian KL partial: -0.5 * sum_h (1 - mu^2 + 2 ls - exp(2 ls))
__global__ void kld_gauss_kernel(const float* __restrict__ mean,
                                 const float* __restrict__ logsigm,
                                 float* __restrict__ kldg) {
  __shared__ float red[256];
  int b = blockIdx.x, tid = threadIdx.x;
  float s = 0.f;
  for (int h = tid; h < Hd; h += blockDim.x) {
    float mu = mean[(size_t)b * Hd + h], ls = logsigm[(size_t)b * Hd + h];
    s += 1.f - mu * mu + 2.f * ls - expf(2.f * ls);
  }
  red[tid] = s; __syncthreads();
  for (int st = blockDim.x / 2; st > 0; st >>= 1) {
    if (tid < st) red[tid] += red[tid + st];
    __syncthreads();
  }
  if (tid == 0) kldg[b] = -0.5f * red[0];
}

__global__ void doc_vec_kernel(const float* __restrict__ mean, const float* __restrict__ logsigm,
                               const float* __restrict__ eps0, float* __restrict__ doc, int n) {
  int i = blockIdx.x * blockDim.x + threadIdx.x;
  if (i < n) doc[i] = expf(logsigm[i]) * eps0[i] + mean[i];
}

// eta = softmax(tanh(eta_h @ eta2_w + eta2_b)) over 3 mixture slots, one thread per b
__global__ void eta_kernel(const float* __restrict__ etah, const float* __restrict__ w,
                           const float* __restrict__ bias, float* __restrict__ eta) {
  int b = blockIdx.x * blockDim.x + threadIdx.x;
  if (b >= Bd) return;
  const float* row = etah + (size_t)b * Hd;
  float s0 = bias[0], s1 = bias[1], s2 = bias[2];
  for (int h = 0; h < Hd; h++) {
    float e = row[h];
    s0 += e * w[h * 3 + 0];
    s1 += e * w[h * 3 + 1];
    s2 += e * w[h * 3 + 2];
  }
  s0 = tanhf(s0); s1 = tanhf(s1); s2 = tanhf(s2);
  float m = fmaxf(s0, fmaxf(s1, s2));
  float e0 = expf(s0 - m), e1 = expf(s1 - m), e2 = expf(s2 - m);
  float inv = 1.f / (e0 + e1 + e2);
  eta[b * 3 + 0] = e0 * inv; eta[b * 3 + 1] = e1 * inv; eta[b * 3 + 2] = e2 * inv;
}

// dep = softmax(depend / temperature, axis=1), one thread per topic row
__global__ void dep_kernel(const float* __restrict__ depend, const int* __restrict__ decay,
                           float* __restrict__ dep) {
  int t = blockIdx.x * blockDim.x + threadIdx.x;
  if (t >= Td) return;
  float invTemp = ((float)decay[0] + 1e-5f) / 0.05f;
  float mx = -INFINITY;
  for (int j = 0; j < T2d; j++) mx = fmaxf(mx, depend[t * T2d + j] * invTemp);
  float s = 0.f;
  for (int j = 0; j < T2d; j++) s += expf(depend[t * T2d + j] * invTemp - mx);
  float inv = 1.f / s;
  for (int j = 0; j < T2d; j++)
    dep[t * T2d + j] = expf(depend[t * T2d + j] * invTemp - mx) * inv;
}

__device__ __forceinline__ float psi_f(float x) {
  float r = 0.f;
  while (x < 6.f) { r -= 1.f / x; x += 1.f; }
  float inv = 1.f / x, inv2 = inv * inv;
  return r + logf(x) - 0.5f * inv
         - inv2 * (1.f / 12.f - inv2 * (1.f / 120.f - inv2 * (1.f / 252.f)));
}

// KL (Kumaraswamy-Beta Taylor) + Kumaraswamy sampling + stick-breaking +
// theta2 = theta@dep, theta3 = sum(theta2) (dep2 softmax over singleton == 1)
__global__ void kl_theta_kernel(const float* __restrict__ a_, const float* __restrict__ b_,
                                const float* __restrict__ eps_u, const float* __restrict__ kldg,
                                const float* __restrict__ dep,
                                float* __restrict__ kld, float* __restrict__ theta,
                                float* __restrict__ theta2, float* __restrict__ theta3) {
  int b = blockIdx.x * blockDim.x + threadIdx.x;
  if (b >= Bd) return;
  float klsum = 0.f, excl = 1.f;
  float th[Td];
  for (int t = 0; t < Td; t++) {
    float av = a_[b * Td + t], bv = b_[b * Td + t];
    float ab = av * bv;
    float kl = 0.f;
#pragma unroll
    for (int m = 1; m <= 10; m++) {
      float p = (float)m / av;
      kl += (1.f / ((float)m + ab)) * expf(lgammaf(p) + lgammaf(bv) - lgammaf(p + bv));
    }
    kl *= (PRIOR_B - 1.f) * bv;
    kl += (av - PRIOR_A) / av * (-0.57721f - psi_f(bv) - 1.f / bv);
    kl += logf(ab) + LOG_BETA_PRIOR;
    kl += -(bv - 1.f) / bv;
    klsum += kl;
    float eps = 0.01f + 0.98f * eps_u[b * Td + t];
    float v = powf(1.f - powf(eps, 1.f / bv), 1.f / av);
    th[t] = v * excl;
    excl *= (1.f - v);
    theta[b * Td + t] = th[t];
  }
  kld[b] = klsum + kldg[b];
  float t3 = 0.f;
  for (int j = 0; j < T2d; j++) {
    float s = 0.f;
    for (int t = 0; t < Td; t++) s += th[t] * dep[t * T2d + j];
    theta2[b * T2d + j] = s;
    t3 += s;
  }
  theta3[b] = t3;
}

// row softmax over V=50000; also emit bf16 copy into bcat row (catOff + r)
__global__ void softmax_row(float* __restrict__ mat, __bf16* __restrict__ bcat, int catOff) {
  __shared__ float red[256];
  int r = blockIdx.x, tid = threadIdx.x, nt = blockDim.x;
  float* row = mat + (size_t)r * Vd;
  __bf16* brow = bcat + (size_t)(catOff + r) * Vd;
  float mx = -INFINITY;
  for (int v = tid; v < Vd; v += nt) mx = fmaxf(mx, row[v]);
  red[tid] = mx; __syncthreads();
  for (int s = nt / 2; s > 0; s >>= 1) {
    if (tid < s) red[tid] = fmaxf(red[tid], red[tid + s]);
    __syncthreads();
  }
  mx = red[0]; __syncthreads();
  float sm = 0.f;
  for (int v = tid; v < Vd; v += nt) sm += expf(row[v] - mx);
  red[tid] = sm; __syncthreads();
  for (int s = nt / 2; s > 0; s >>= 1) {
    if (tid < s) red[tid] += red[tid + s];
    __syncthreads();
  }
  float inv = 1.f / red[0];
  for (int v = tid; v < Vd; v += nt) {
    float p = expf(row[v] - mx) * inv;
    row[v] = p;
    brow[v] = (__bf16)p;
  }
}

// zero K-pad rows 81..95 of bcat
__global__ void pad_bcat(__bf16* __restrict__ bcat) {
  size_t idx = (size_t)blockIdx.x * blockDim.x + threadIdx.x;
  if (idx < (size_t)(KCAT - (Td + T2d + 1)) * Vd)
    bcat[(size_t)(Td + T2d + 1) * Vd + idx] = (__bf16)0.f;
}

// A_cat[b, 0..95] = [eta0*theta | eta1*theta2 | eta2*theta3 | zeros]
__global__ void acat_kernel(const float* __restrict__ eta, const float* __restrict__ theta,
                            const float* __restrict__ theta2, const float* __restrict__ theta3,
                            __bf16* __restrict__ Acat) {
  int idx = blockIdx.x * blockDim.x + threadIdx.x;
  if (idx >= Bd * KCAT) return;
  int b = idx / KCAT, c = idx % KCAT;
  float v;
  if (c < Td)             v = eta[b * 3 + 0] * theta[b * Td + c];
  else if (c < Td + T2d)  v = eta[b * 3 + 1] * theta2[b * T2d + (c - Td)];
  else if (c == Td + T2d) v = eta[b * 3 + 2] * theta3[b];
  else v = 0.f;
  Acat[idx] = (__bf16)v;
}

// fused: part[gy][b] = sum over this block's v-range of log(A_cat[b,:]@B_cat[:,v]) * x[b,v]
// All 3 B fragments loaded before the 3 back-to-back WMMAs; next tile prefetched.
__global__ void fused_recon(const __bf16* __restrict__ Acat, const __bf16* __restrict__ bcat,
                            const float* __restrict__ x, float* __restrict__ part,
                            int tilesPerBlock) {
  const int lane = (int)threadIdx.x;
  const int b0 = blockIdx.x * 16;
  const int rc = lane & 15;
  const int halfOff = (lane < 16) ? 0 : 8;

  // preload A fragments for the 3 k-steps (K = 96)
  v16bf af[3];
  const __bf16* arow = Acat + (size_t)(b0 + rc) * KCAT;
#pragma unroll
  for (int ks = 0; ks < 3; ks++) {
    int base = ks * 32 + halfOff;
#pragma unroll
    for (int e = 0; e < 8; e++) af[ks][e]     = arow[base + e];
#pragma unroll
    for (int e = 0; e < 8; e++) af[ks][8 + e] = arow[base + 16 + e];
  }

  float rsum[8];
#pragma unroll
  for (int r = 0; r < 8; r++) rsum[r] = 0.f;

  const int vtile0 = blockIdx.y * tilesPerBlock;
  for (int i = 0; i < tilesPerBlock; i++) {
    const int v0 = (vtile0 + i) * 16;
    // ---- issue all 3 B-fragment loads ----
    v16bf bf[3];
#pragma unroll
    for (int ks = 0; ks < 3; ks++) {
      const __bf16* bp = bcat + (size_t)(ks * 32 + halfOff) * Vd + v0 + rc;
#pragma unroll
      for (int e = 0; e < 16; e++) bf[ks][e] = bp[(size_t)(e < 8 ? e : e + 8) * Vd];
    }
    // prefetch next tile's B rows
    if (i + 1 < tilesPerBlock) {
      __builtin_prefetch(bcat + (size_t)lane * Vd + v0 + 16, 0, 1);
      __builtin_prefetch(bcat + (size_t)(32 + lane) * Vd + v0 + 16, 0, 1);
      __builtin_prefetch(bcat + (size_t)(64 + lane) * Vd + v0 + 16, 0, 1);
    }
    // ---- 3 chained WMMAs ----
    v8f acc = {};
#pragma unroll
    for (int ks = 0; ks < 3; ks++)
      acc = __builtin_amdgcn_wmma_f32_16x16x32_bf16(false, af[ks], false, bf[ks],
                                                    (short)0, acc, false, false);
    // ---- epilogue: log(.) * x ----
#pragma unroll
    for (int r = 0; r < 8; r++) {
      const int m = b0 + r + ((lane < 16) ? 0 : 8);
      rsum[r] += logf(acc[r]) * x[(size_t)m * Vd + (v0 + rc)];
    }
  }
  // reduce over the 16 N-lanes in each half, write per-(gy,b) partial (deterministic)
#pragma unroll
  for (int r = 0; r < 8; r++) {
    float s = rsum[r];
    s += __shfl_xor(s, 1);
    s += __shfl_xor(s, 2);
    s += __shfl_xor(s, 4);
    s += __shfl_xor(s, 8);
    if (rc == 0) {
      const int m = b0 + r + ((lane < 16) ? 0 : 8);
      part[(size_t)blockIdx.y * Bd + m] = s;
    }
  }
}

// recons[b] = -sum_gy part[gy][b]  (fixed order -> deterministic)
__global__ void recon_reduce(const float* __restrict__ part, float* __restrict__ recons) {
  int b = blockIdx.x * blockDim.x + threadIdx.x;
  if (b >= Bd) return;
  float s = 0.f;
  for (int y = 0; y < RECON_GY; y++) s += part[(size_t)y * Bd + b];
  recons[b] = -s;
}

// ---------------- launcher ----------------
extern "C" void kernel_launch(void* const* d_in, const int* in_sizes, int n_in,
                              void* d_out, int out_size, void* d_ws, size_t ws_size,
                              hipStream_t stream) {
  const float* x          = (const float*)d_in[0];
  const float* eps0       = (const float*)d_in[1];
  const float* eps_u      = (const float*)d_in[2];
  const float* mlp_w      = (const float*)d_in[3];
  const float* mlp_b      = (const float*)d_in[4];
  const float* a_w        = (const float*)d_in[5];
  const float* a_b        = (const float*)d_in[6];
  const float* b_w        = (const float*)d_in[7];
  const float* b_b        = (const float*)d_in[8];
  const float* mean_w     = (const float*)d_in[9];
  const float* mean_b     = (const float*)d_in[10];
  const float* logs_w     = (const float*)d_in[11];
  const float* logs_b     = (const float*)d_in[12];
  const float* eta1_w     = (const float*)d_in[13];
  const float* eta1_b     = (const float*)d_in[14];
  const float* eta2_w     = (const float*)d_in[15];
  const float* eta2_b     = (const float*)d_in[16];
  const float* word_vec   = (const float*)d_in[17];
  const float* topic_vec  = (const float*)d_in[18];
  const float* depend     = (const float*)d_in[19];
  const float* topic_vec2 = (const float*)d_in[20];
  // d_in[21] = depend2: softmax over axis of size 1 == 1.0, value unused
  const float* topic_vec3 = (const float*)d_in[22];
  const int*   decay      = (const int*)d_in[23];

  // -------- output layout (concat, fp32) --------
  float* out = (float*)d_out;
  float* o_recons = out;                         // [B]
  float* o_kld    = o_recons + Bd;               // [B]
  float* o_theta  = o_kld + Bd;                  // [B,T]
  float* o_theta2 = o_theta + (size_t)Bd * Td;   // [B,T2]
  float* o_theta3 = o_theta2 + (size_t)Bd * T2d; // [B]
  float* o_beta   = o_theta3 + Bd;               // [T,V]
  float* o_beta2  = o_beta + (size_t)Td * Vd;    // [T2,V]
  float* o_beta3  = o_beta2 + (size_t)T2d * Vd;  // [1,V]

  // -------- workspace carve --------
  char* w = (char*)d_ws;
  auto alloc = [&](size_t bytes) -> void* {
    void* p = (void*)w;
    w += (bytes + 255) & ~(size_t)255;
    return p;
  };
  __bf16* wvT   = (__bf16*)alloc((size_t)Hd * Vd * sizeof(__bf16));      // word_vec^T bf16 [H,V]
  float*  partK = (float*)alloc((size_t)SPLIT * Bd * Hd * 4);            // split-K partials
  float*  partR = (float*)alloc((size_t)RECON_GY * Bd * 4);              // recon partials
  float*  enc   = (float*)alloc((size_t)Bd * Hd * 4);
  float*  meanb = (float*)alloc((size_t)Bd * Hd * 4);
  float*  logsb = (float*)alloc((size_t)Bd * Hd * 4);
  float*  doc   = (float*)alloc((size_t)Bd * Hd * 4);
  float*  etah  = (float*)alloc((size_t)Bd * Hd * 4);
  float*  abuf  = (float*)alloc((size_t)Bd * Td * 4);
  float*  bbuf  = (float*)alloc((size_t)Bd * Td * 4);
  float*  eta   = (float*)alloc((size_t)Bd * 3 * 4);
  float*  dep   = (float*)alloc((size_t)Td * T2d * 4);
  float*  kldg  = (float*)alloc((size_t)Bd * 4);
  __bf16* Acat  = (__bf16*)alloc((size_t)Bd * KCAT * sizeof(__bf16));
  __bf16* bcat  = (__bf16*)alloc((size_t)KCAT * Vd * sizeof(__bf16));    // [96,V]

  // ---- transpose word_vec [V,H] -> bf16 [H,V] ----
  {
    size_t tot = (size_t)Hd * Vd;
    wv_transpose<<<(unsigned)((tot + 255) / 256), 256, 0, stream>>>(word_vec, wvT);
  }

  // ---- GEMM1 (26 GFLOP): split-K=8 WMMA, deterministic reduce + bias + tanh ----
  {
    const int kIters = (Vd + 31) / 32;                  // 1563
    const int kPerSlice = (kIters + SPLIT - 1) / SPLIT; // 196
    wmma_gemm<0, false, true><<<dim3(Hd / 64, Bd / 16, SPLIT), 32, 0, stream>>>(
        x, mlp_w, nullptr, partK, Bd, Hd, Vd, kPerSlice);
    splitk_reduce<1><<<(Bd * Hd + 255) / 256, 256, 0, stream>>>(
        partK, mlp_b, enc, Bd * Hd, Hd, SPLIT);
  }

  // ---- encoder head GEMMs (WMMA bf16, fused bias + activation) ----
  wmma_gemm<2, false, false><<<dim3(1, Bd / 16), 32, 0, stream>>>(enc, a_w, a_b, abuf, Bd, Td, Hd, 0);
  wmma_gemm<2, false, false><<<dim3(1, Bd / 16), 32, 0, stream>>>(enc, b_w, b_b, bbuf, Bd, Td, Hd, 0);
  wmma_gemm<0, false, false><<<dim3(Hd / 64, Bd / 16), 32, 0, stream>>>(enc, mean_w, mean_b, meanb, Bd, Hd, Hd, 0);
  wmma_gemm<0, false, false><<<dim3(Hd / 64, Bd / 16), 32, 0, stream>>>(enc, logs_w, logs_b, logsb, Bd, Hd, Hd, 0);

  kld_gauss_kernel<<<Bd, 256, 0, stream>>>(meanb, logsb, kldg);
  doc_vec_kernel<<<(Bd * Hd + 255) / 256, 256, 0, stream>>>(meanb, logsb, eps0, doc, Bd * Hd);
  wmma_gemm<1, false, false><<<dim3(Hd / 64, Bd / 16), 32, 0, stream>>>(doc, eta1_w, eta1_b, etah, Bd, Hd, Hd, 0);
  eta_kernel<<<(Bd + 63) / 64, 64, 0, stream>>>(etah, eta2_w, eta2_b, eta);

  // ---- sampling / stick-breaking / KL ----
  dep_kernel<<<2, 32, 0, stream>>>(depend, decay, dep);
  kl_theta_kernel<<<(Bd + 63) / 64, 64, 0, stream>>>(abuf, bbuf, eps_u, kldg, dep,
                                                     o_kld, o_theta, o_theta2, o_theta3);

  // ---- topic-word distributions: scores (WMMA) then row softmax ----
  {
    const int nStrips = (Vd + 63) / 64; // 782, last strip has 1 valid 16-tile
    wmma_gemm<0, true, false><<<dim3(nStrips, Td / 16), 32, 0, stream>>>(topic_vec, wvT, nullptr, o_beta, Td, Vd, Hd, 0);
    wmma_gemm<0, true, false><<<dim3(nStrips, 1), 32, 0, stream>>>(topic_vec2, wvT, nullptr, o_beta2, T2d, Vd, Hd, 0);
    wmma_gemm<0, true, false><<<dim3(nStrips, 1), 32, 0, stream>>>(topic_vec3, wvT, nullptr, o_beta3, 1, Vd, Hd, 0);
  }
  softmax_row<<<Td, 256, 0, stream>>>(o_beta, bcat, 0);
  softmax_row<<<T2d, 256, 0, stream>>>(o_beta2, bcat, Td);
  softmax_row<<<1, 256, 0, stream>>>(o_beta3, bcat, Td + T2d);
  pad_bcat<<<((KCAT - (Td + T2d + 1)) * Vd + 255) / 256, 256, 0, stream>>>(bcat);

  // ---- fused reconstruction loss (WMMA + log epilogue, deterministic reduce) ----
  acat_kernel<<<(Bd * KCAT + 255) / 256, 256, 0, stream>>>(eta, o_theta, o_theta2, o_theta3, Acat);
  fused_recon<<<dim3(Bd / 16, RECON_GY), 32, 0, stream>>>(Acat, bcat, x, partR, 25);
  recon_reduce<<<(Bd + 255) / 256, 256, 0, stream>>>(partR, o_recons);
}
